// ConvBlock_1614907703435
// MI455X (gfx1250) — matrix operand order repaired and checked by reading, not compiled, not run on hardware
//
#include <hip/hip_runtime.h>
#include <math.h>

typedef __attribute__((ext_vector_type(16))) _Float16 v16h;
typedef __attribute__((ext_vector_type(8)))  _Float16 v8h;
typedef __attribute__((ext_vector_type(8)))  float    v8f;
typedef __attribute__((ext_vector_type(4)))  unsigned int u32x4;
typedef __attribute__((ext_vector_type(8)))  int i32x8;
typedef __attribute__((ext_vector_type(4)))  int i32x4;

#define B_  8
#define C_  512
#define CH_ 2048
#define K_  5
#define L_  2048
#define CS_ 16
#define T_  128   // L / CS
#define C8_ 64
#define LP_ (L_ + 16)   // padded length in inH (16 zero rows of history)

#define LT  64          // output columns per tile
#define XT  80          // extended columns (LT + 16 history for causal conv)
#define CHUNK 128       // CH rows processed per chunk
#define NCHUNK (CH_/CHUNK)

#define IN_STRIDE  (C_+8)      // 520 halves; matches TDM pad: 1024B interval + 16B pad
#define X1_STRIDE  (XT+8)      // 88
#define HT_STRIDE  (CHUNK+8)   // 136

__device__ __forceinline__ v16h pack16(v8h lo, v8h hi) {
  union { v16h v; v8h h[2]; } u;
  u.h[0] = lo; u.h[1] = hi;
  return u.v;
}

__device__ __forceinline__ float wave_reduce_add(float v) {
  #pragma unroll
  for (int m = 16; m > 0; m >>= 1) v += __shfl_xor(v, m, 32);
  return v;
}

// ---------------- weight prep: per-row stdize (ddof=1) ----------------

__global__ void stdize_rows_f16_kernel(const float* __restrict__ w,
                                       const float* __restrict__ s,
                                       _Float16* __restrict__ out, int cols) {
  const int wid = threadIdx.x >> 5, lane = threadIdx.x & 31;
  const int row = blockIdx.x * 8 + wid;
  const float* wr = w + (size_t)row * cols;
  float sum = 0.0f, sq = 0.0f;
  for (int i = lane; i < cols; i += 32) { float v = wr[i]; sum += v; sq += v * v; }
  sum = wave_reduce_add(sum); sq = wave_reduce_add(sq);
  const float n = (float)cols;
  const float var = (sq - sum * sum / n) / (n - 1.0f);
  const float scale = s[row] * rsqrtf(var);
  _Float16* orow = out + (size_t)row * cols;
  for (int i = lane; i < cols; i += 32) orow[i] = (_Float16)(wr[i] * scale);
}

__global__ void stdize_rows_f32_kernel(const float* __restrict__ w,
                                       const float* __restrict__ s,
                                       float* __restrict__ out, int cols) {
  const int wid = threadIdx.x >> 5, lane = threadIdx.x & 31;
  const int row = blockIdx.x * 8 + wid;
  const float* wr = w + (size_t)row * cols;
  float sum = 0.0f, sq = 0.0f;
  for (int i = lane; i < cols; i += 32) { float v = wr[i]; sum += v; sq += v * v; }
  sum = wave_reduce_add(sum); sq = wave_reduce_add(sq);
  const float n = (float)cols;
  const float var = (sq - sum * sum / n) / (n - 1.0f);
  const float scale = s[row] * rsqrtf(var);
  float* orow = out + (size_t)row * cols;
  for (int i = lane; i < cols; i += 32) orow[i] = wr[i] * scale;
}

__global__ void stdize_wd_kernel(const float* __restrict__ wd,
                                 const float* __restrict__ sd,
                                 float* __restrict__ out) {
  const int r = blockIdx.x * 256 + threadIdx.x;
  if (r >= CH_) return;
  const float* wr = wd + r * K_;
  float sum = 0.0f, sq = 0.0f;
  #pragma unroll
  for (int k = 0; k < K_; ++k) { float v = wr[k]; sum += v; sq += v * v; }
  const float var = (sq - sum * sum / (float)K_) / ((float)K_ - 1.0f);
  const float scale = sd[r] * rsqrtf(var);
  #pragma unroll
  for (int k = 0; k < K_; ++k) out[r * K_ + k] = wr[k] * scale;
}

// ------- input prep: [B,C,L] f32 -> [B, 16+L, C] f16 (transposed, zero head) -------

__global__ void zero_head_kernel(_Float16* __restrict__ inH) {
  const int i = blockIdx.x * 256 + threadIdx.x;     // over B*16*C
  const int b = i / (16 * C_);
  const int r = i % (16 * C_);
  inH[(size_t)b * LP_ * C_ + r] = (_Float16)0.0f;
}

__global__ void in_transpose_kernel(const float* __restrict__ in,
                                    _Float16* __restrict__ inH) {
  __shared__ _Float16 tile[64][64 + 8];
  const int b  = blockIdx.z;
  const int c0 = blockIdx.y * 64;
  const int l0 = blockIdx.x * 64;
  const int tx = threadIdx.x & 63;
  const int ty = threadIdx.x >> 6;   // 0..3
  #pragma unroll
  for (int i = 0; i < 64; i += 4) {
    const int c = c0 + ty + i;
    tile[ty + i][tx] = (_Float16)in[(size_t)(b * C_ + c) * L_ + l0 + tx];
  }
  __syncthreads();
  #pragma unroll
  for (int i = 0; i < 64; i += 4) {
    const int l = l0 + ty + i;
    inH[((size_t)b * LP_ + 16 + l) * C_ + c0 + tx] = tile[tx][ty + i];
  }
}

// ---------- fused pw1 -> depthwise causal conv -> pw2 (WMMA f16 + TDM) ----------

__global__ void __launch_bounds__(256)
fused_conv_kernel(const _Float16* __restrict__ inH,  // [B, 16+L, C] f16
                  const _Float16* __restrict__ w1h,  // [CH, C] f16 stdized
                  const float*    __restrict__ b1,
                  const float*    __restrict__ wdf,  // [CH, K] f32 stdized
                  const float*    __restrict__ bd,
                  const _Float16* __restrict__ w2h,  // [C, CH] f16 stdized
                  float*          __restrict__ x2)   // [B, C, L] f32
{
  extern __shared__ char smem_raw[];
  _Float16* inT = (_Float16*)smem_raw;                 // [XT][IN_STRIDE]
  _Float16* x1b = inT + XT * IN_STRIDE;                // [CHUNK][X1_STRIDE]
  _Float16* hT  = x1b + CHUNK * X1_STRIDE;             // [LT][HT_STRIDE]

  const int tid  = threadIdx.x;
  const int wid  = tid >> 5;
  const int lane = tid & 31;
  const int lr   = lane & 15;
  const int hi   = lane >> 4;
  const int b    = blockIdx.y;
  const int l0   = blockIdx.x * LT;

  // ---- TDM: DMA the 80x512 f16 tile into LDS (pad reproduces IN_STRIDE=520) ----
  if (wid == 0) {
    const unsigned lds_base = (unsigned)(size_t)(void*)inT;
    const unsigned long long ga =
        (unsigned long long)(size_t)(inH + ((size_t)(b * LP_ + l0) * C_));
    u32x4 g0;
    g0[0] = 1u;                                                 // count=1 (valid)
    g0[1] = lds_base;                                           // lds_addr
    g0[2] = (unsigned)ga;                                       // global_addr[31:0]
    g0[3] = (unsigned)((ga >> 32) & 0x1FFFFFFu) | (2u << 30);   // addr[56:32], type=2
    i32x8 g1;
    g1[0] = (1 << 16)      // data_size = 2 bytes
          | (1 << 20)      // pad_enable
          | (7 << 22)      // pad_interval: 256 DWORDs = 1024 B
          | (3 << 25);     // pad_amount: 4 DWORDs = 16 B
    g1[1] = (int)(512u << 16);           // tensor_dim0 = 512 (bits 79:48)
    g1[2] = (int)((unsigned)LP_ << 16);  // tensor_dim1 low bits (bits 111:80)
    g1[3] = (int)(512u << 16);           // tile_dim0 = 512 (bits 127:112)
    g1[4] = XT;                          // tile_dim1 = 80 (bits 143:128)
    g1[5] = 512;                         // tensor_dim0_stride = 512 (bits 207:160)
    g1[6] = 0;
    g1[7] = 0;
    const i32x4 gz4 = {0, 0, 0, 0};
    const i32x8 gz8 = {0, 0, 0, 0, 0, 0, 0, 0};
    __builtin_amdgcn_tensor_load_to_lds(g0, g1, gz4, gz4, gz8, 0);
    __builtin_amdgcn_s_wait_tensorcnt(0);
  }
  __syncthreads();

  v8f acc2[16] = {};   // persistent 512x64 f32 tile: wave owns 4 row-tiles x 4 col-tiles

  for (int ci = 0; ci < NCHUNK; ++ci) {
    const int ch0 = ci * CHUNK;

    // ---- GEMM1: x1 = relu(W1 @ in + b1) for rows [ch0+16*wid, +16), cols [0,80)
    v8f acc1[5] = {};
    const int arow = ch0 + wid * 16 + lr;
    const _Float16* aptr = w1h + (size_t)arow * C_;
    #pragma unroll 4
    for (int kk = 0; kk < C_ / 32; ++kk) {
      const int koffA = kk * 32 + hi * 8;
      v16h a = pack16(*(const v8h*)(aptr + koffA),
                      *(const v8h*)(aptr + koffA + 16));
      const int koffB = kk * 32 + hi * 8;
      #pragma unroll
      for (int ct = 0; ct < 5; ++ct) {
        const _Float16* bp = inT + (ct * 16 + lr) * IN_STRIDE + koffB;
        v16h bb = pack16(*(const v8h*)bp, *(const v8h*)(bp + 16));
        acc1[ct] = __builtin_amdgcn_wmma_f32_16x16x32_f16(
            false, a, false, bb, (short)0, acc1[ct], false, false);
      }
    }
    // bias + relu; reference zero-pads POST-relu x1 left of l=0, so mask here
    #pragma unroll
    for (int v = 0; v < 8; ++v) {
      const int rloc = wid * 16 + v + 8 * hi;
      const float bias = b1[ch0 + rloc];
      #pragma unroll
      for (int ct = 0; ct < 5; ++ct) {
        const int col = ct * 16 + lr;
        float x = fmaxf(acc1[ct][v] + bias, 0.0f);
        if (l0 - 16 + col < 0) x = 0.0f;
        x1b[rloc * X1_STRIDE + col] = (_Float16)x;
      }
    }
    __syncthreads();

    // ---- depthwise causal conv + relu, stored transposed (CH contiguous)
    for (int idx = tid; idx < CHUNK * LT; idx += 256) {
      const int r = idx >> 6;    // chunk-local channel
      const int j = idx & 63;    // output column
      const int ch = ch0 + r;
      const _Float16* xr = x1b + r * X1_STRIDE + 12 + j;  // cols l-4 .. l
      const float* wr = wdf + ch * K_;
      float acc = bd[ch];
      #pragma unroll
      for (int k = 0; k < K_; ++k) acc += wr[k] * (float)xr[k];
      hT[j * HT_STRIDE + r] = (_Float16)fmaxf(acc, 0.0f);
    }
    __syncthreads();

    // ---- GEMM2: acc2 += W2[:, ch0:ch0+128] @ h
    #pragma unroll
    for (int kk = 0; kk < CHUNK / 32; ++kk) {
      const int koffA = kk * 32 + hi * 8;
      const int koffB = kk * 32 + hi * 8;
      #pragma unroll
      for (int rt = 0; rt < 4; ++rt) {
        const int row = (wid * 4 + rt) * 16 + lr;
        const _Float16* ap = w2h + (size_t)row * CH_ + ch0 + koffA;
        v16h a = pack16(*(const v8h*)ap, *(const v8h*)(ap + 16));
        #pragma unroll
        for (int ct = 0; ct < 4; ++ct) {
          const _Float16* bp = hT + (ct * 16 + lr) * HT_STRIDE + koffB;
          v16h bb = pack16(*(const v8h*)bp, *(const v8h*)(bp + 16));
          acc2[rt * 4 + ct] = __builtin_amdgcn_wmma_f32_16x16x32_f16(
              false, a, false, bb, (short)0, acc2[rt * 4 + ct], false, false);
        }
      }
    }
    __syncthreads();
  }

  // write x2 tile (f32, full precision for SE + residual)
  #pragma unroll
  for (int rt = 0; rt < 4; ++rt) {
    #pragma unroll
    for (int v = 0; v < 8; ++v) {
      const int c = (wid * 4 + rt) * 16 + v + 8 * hi;
      #pragma unroll
      for (int ct = 0; ct < 4; ++ct) {
        const int l = l0 + ct * 16 + lr;
        x2[(size_t)(b * C_ + c) * L_ + l] = acc2[rt * 4 + ct][v];
      }
    }
  }
}

// ---------------- SE: pool over CS, per-channel EMA ----------------

__global__ void pool_ema_kernel(const float* __restrict__ x2,
                                const float* __restrict__ ema_r,
                                float* __restrict__ e) {
  __shared__ float pbuf[8][T_];
  const int wid = threadIdx.x >> 5, lane = threadIdx.x & 31;
  const int idx = blockIdx.x * 8 + wid;
  const int b = idx / C_, c = idx % C_;
  const float* row = x2 + (size_t)(b * C_ + c) * L_ + lane * 64;
  #pragma unroll
  for (int g = 0; g < 4; ++g) {
    float s = 0.0f;
    #pragma unroll
    for (int i = 0; i < CS_; ++i) s += row[g * CS_ + i];
    pbuf[wid][lane * 4 + g] = s * (1.0f / CS_);
  }
  __syncthreads();
  if (lane == 0) {
    const float r = ema_r[c];
    float y = 0.0f;
    float* erow = e + (size_t)(b * C_ + c) * T_;
    for (int t = 0; t < T_; ++t) {
      y = r * y + (1.0f - r) * pbuf[wid][t];
      erow[t] = y;
    }
  }
}

// ---------------- SE MLP: C -> C8 relu -> C, sigmoid gate ----------------

__global__ void se_mlp_kernel(const float* __restrict__ e,
                              const float* __restrict__ wse1, const float* __restrict__ bse1,
                              const float* __restrict__ wse2, const float* __restrict__ bse2,
                              const float* __restrict__ se_scale,
                              float* __restrict__ g) {
  __shared__ float es[C_];
  __shared__ float h1[C8_];
  const int bt = blockIdx.x;            // b*T + t
  const int b = bt / T_, t = bt % T_;
  const int tid = threadIdx.x;
  for (int c = tid; c < C_; c += 256) es[c] = e[(size_t)(b * C_ + c) * T_ + t];
  __syncthreads();
  if (tid < C8_) {
    const float* wr = wse1 + tid * C_;
    float acc = bse1[tid];
    for (int c = 0; c < C_; ++c) acc += wr[c] * es[c];
    h1[tid] = fmaxf(acc, 0.0f);
  }
  __syncthreads();
  for (int c = tid; c < C_; c += 256) {
    const float* wr = wse2 + c * C8_;
    float acc = bse2[c];
    #pragma unroll
    for (int k = 0; k < C8_; ++k) acc += wr[k] * h1[k];
    acc *= se_scale[c];
    g[(size_t)(b * C_ + c) * T_ + t] = 1.0f / (1.0f + expf(-acc));
  }
}

// ---------------- final: out = inputs + x2 * repeat(g, CS) ----------------

__global__ void final_kernel(const float* __restrict__ in,
                             const float* __restrict__ x2,
                             const float* __restrict__ g,
                             float* __restrict__ out) {
  const size_t i = ((size_t)blockIdx.x * 256 + threadIdx.x) * 4;
  const int l = (int)(i % L_);
  const size_t bc = i / L_;
  const float gv = g[bc * T_ + (l >> 4)];
  const float4 a = *(const float4*)(in + i);
  const float4 x = *(const float4*)(x2 + i);
  float4 o;
  o.x = a.x + x.x * gv; o.y = a.y + x.y * gv;
  o.z = a.z + x.z * gv; o.w = a.w + x.w * gv;
  *(float4*)(out + i) = o;
}

__global__ void copy_lengths_kernel(const long long* __restrict__ len,
                                    long long* __restrict__ dst) {
  if (threadIdx.x < B_) dst[threadIdx.x] = len[threadIdx.x];
}

// ---------------- launch ----------------

extern "C" void kernel_launch(void* const* d_in, const int* in_sizes, int n_in,
                              void* d_out, int out_size, void* d_ws, size_t ws_size,
                              hipStream_t stream) {
  (void)in_sizes; (void)n_in; (void)out_size; (void)ws_size;
  const float*     inputs  = (const float*)d_in[0];
  const long long* lengths = (const long long*)d_in[1];
  const float* w1 = (const float*)d_in[3];
  const float* b1 = (const float*)d_in[4];
  const float* s1 = (const float*)d_in[5];
  const float* wd = (const float*)d_in[6];
  const float* bd = (const float*)d_in[7];
  const float* sd = (const float*)d_in[8];
  const float* w2 = (const float*)d_in[9];
  const float* s2 = (const float*)d_in[10];
  const float* ema_r = (const float*)d_in[11];
  const float* wse1 = (const float*)d_in[12];
  const float* bse1 = (const float*)d_in[13];
  const float* sse1 = (const float*)d_in[14];
  const float* wse2 = (const float*)d_in[15];
  const float* bse2 = (const float*)d_in[16];
  const float* sse2 = (const float*)d_in[17];
  const float* se_scale = (const float*)d_in[18];

  char* ws = (char*)d_ws;
  float*    x2    = (float*)(ws);                     // 33,554,432 B
  float*    e     = (float*)(ws + 33554432);          //  2,097,152 B
  float*    g     = (float*)(ws + 35651584);          //  2,097,152 B
  _Float16* w1h   = (_Float16*)(ws + 37748736);       //  2,097,152 B
  _Float16* w2h   = (_Float16*)(ws + 39845888);       //  2,097,152 B
  float*    wdf   = (float*)(ws + 41943040);          //     40,960 B
  float*    wse1f = (float*)(ws + 41984000);          //    131,072 B
  float*    wse2f = (float*)(ws + 42115072);          //    131,072 B
  _Float16* inH   = (_Float16*)(ws + 42246144);       // 16,908,288 B  (B * 2064 * 512 f16)

  stdize_rows_f16_kernel<<<CH_ / 8, 256, 0, stream>>>(w1, s1, w1h, C_);
  stdize_rows_f16_kernel<<<C_ / 8, 256, 0, stream>>>(w2, s2, w2h, CH_);
  stdize_rows_f32_kernel<<<C8_ / 8, 256, 0, stream>>>(wse1, sse1, wse1f, C_);
  stdize_rows_f32_kernel<<<C_ / 8, 256, 0, stream>>>(wse2, sse2, wse2f, C8_);
  stdize_wd_kernel<<<CH_ / 256, 256, 0, stream>>>(wd, sd, wdf);

  zero_head_kernel<<<(B_ * 16 * C_) / 256, 256, 0, stream>>>(inH);
  in_transpose_kernel<<<dim3(L_ / 64, C_ / 64, B_), 256, 0, stream>>>(inputs, inH);

  dim3 gridA(L_ / LT, B_);
  const size_t smemA =
      (size_t)(XT * IN_STRIDE + CHUNK * X1_STRIDE + LT * HT_STRIDE) * sizeof(_Float16);
  fused_conv_kernel<<<gridA, 256, smemA, stream>>>(inH, w1h, b1, wdf, bd, w2h, x2);

  pool_ema_kernel<<<(B_ * C_) / 8, 256, 0, stream>>>(x2, ema_r, e);
  se_mlp_kernel<<<B_ * T_, 256, 0, stream>>>(e, wse1f, bse1, wse2f, bse2, se_scale, g);
  final_kernel<<<(B_ * C_ * L_) / (256 * 4), 256, 0, stream>>>(inputs, x2, g, (float*)d_out);
  copy_lengths_kernel<<<1, 32, 0, stream>>>(
      lengths, (long long*)((char*)d_out + (size_t)B_ * C_ * L_ * sizeof(float)));
}
